// AggAttn_8495445312090
// MI455X (gfx1250) — matrix operand order repaired and checked by reading, not compiled, not run on hardware
//
#include <hip/hip_runtime.h>
#include <hip/hip_bf16.h>
#include <math.h>

#define D_MODEL   1024
#define NUM_HEADS 16
#define HEAD_DIM  64
#define BATCH     4
#define SEQ       1024

typedef __attribute__((ext_vector_type(16))) __bf16 v16bf;
typedef __attribute__((ext_vector_type(8)))  __bf16 v8bf;
typedef __attribute__((ext_vector_type(8)))  float  v8f;

// ---------------- WMMA fragment helpers (wave32, 16x16x32 bf16) ----------------
// A-matrix 16x32 bf16 layout (ISA 7.12.2): lanes 0-15 -> M=lane, K={0..7,16..23};
// lanes 16-31 -> M=lane-16, K={8..15,24..31}. Two 16B chunks per lane.
__device__ __forceinline__ v16bf load_a_bf16(const __bf16* base, int ld, int row0, int k0) {
  const int lane = threadIdx.x & 31;
  const __bf16* p = base + (size_t)(row0 + (lane & 15)) * ld + k0 + ((lane >> 4) << 3);
  v8bf lo = *(const v8bf*)p;
  v8bf hi = *(const v8bf*)(p + 16);
  v16bf f;
#pragma unroll
  for (int i = 0; i < 8; ++i) { f[i] = lo[i]; f[i + 8] = hi[i]; }
  return f;
}

// Same A layout, source f32 (attention probabilities) converted on the fly.
__device__ __forceinline__ v16bf load_a_f32(const float* base, int ld, int row0, int k0) {
  const int lane = threadIdx.x & 31;
  const float* p = base + (size_t)(row0 + (lane & 15)) * ld + k0 + ((lane >> 4) << 3);
  v16bf f;
#pragma unroll
  for (int i = 0; i < 8; ++i) { f[i] = (__bf16)p[i]; f[i + 8] = (__bf16)p[i + 16]; }
  return f;
}

// B-matrix 32x16 bf16 layout: lanes 0-15 -> N=lane, K={0..7,16..23}; lanes 16-31 -> N=lane-16,
// K={8..15,24..31}. baseT is B^T stored row-major [N x K] (K contiguous) -> two 16B chunks per lane.
__device__ __forceinline__ v16bf load_b_bf16(const __bf16* baseT, int ld, int col0, int k0) {
  const int lane = threadIdx.x & 31;
  const __bf16* p = baseT + (size_t)(col0 + (lane & 15)) * ld + k0 + ((lane >> 4) << 3);
  v8bf lo = *(const v8bf*)p;
  v8bf hi = *(const v8bf*)(p + 16);
  v16bf f;
#pragma unroll
  for (int i = 0; i < 8; ++i) { f[i] = lo[i]; f[i + 8] = hi[i]; }
  return f;
}

// C/D f32 16x16 layout: VGPR r: lanes 0-15 -> (M=r, N=lane), lanes 16-31 -> (M=8+r, N=lane-16)
__device__ __forceinline__ void store_c_f32(float* base, int ld, int row0, int col0, v8f c) {
  const int lane = threadIdx.x & 31;
  const int col = col0 + (lane & 15);
  const int r0  = row0 + ((lane >> 4) << 3);
#pragma unroll
  for (int r = 0; r < 8; ++r) base[(size_t)(r0 + r) * ld + col] = c[r];
}

__device__ __forceinline__ void store_c_bf16(__bf16* base, int ld, int row0, int col0, v8f c) {
  const int lane = threadIdx.x & 31;
  const int col = col0 + (lane & 15);
  const int r0  = row0 + ((lane >> 4) << 3);
#pragma unroll
  for (int r = 0; r < 8; ++r) base[(size_t)(r0 + r) * ld + col] = (__bf16)c[r];
}

__device__ __forceinline__ v8f wmma_bf16(v16bf a, v16bf b, v8f c) {
  return __builtin_amdgcn_wmma_f32_16x16x32_bf16(false, a, false, b, (short)0, c, false, false);
}

// ---------------- Kernels ----------------

__global__ void aggattn_lambda(const float* lq, const float* lk, const float* lv, float* lam) {
  if (threadIdx.x == 0) {
    float s = 0.f, c = 0.f;
    for (int i = 0; i < HEAD_DIM; ++i) { s += lq[i] * lk[i]; c += lk[i] * lv[i]; }
    lam[0] = __expf(s);   // lam_self
    lam[1] = __expf(c);   // lam_cross
  }
}

__global__ void aggattn_cvt(const float* __restrict__ in, __bf16* __restrict__ out, int n) {
  int i = blockIdx.x * blockDim.x + threadIdx.x;
  if (i < n) out[i] = (__bf16)in[i];
}

// Blocked GEMM: C = A[MxK] @ B^T (Bt row-major [NxK]) + bias[N].
// One wave -> 64x64 output (4x4 tiles, 16 accumulators); 16 WMMAs per 8 fragment loads.
// launch_bounds(256,1): allow ~224 VGPRs/lane so accumulators never spill.
template <bool OUT_BF16>
__global__ void __launch_bounds__(256, 1)
aggattn_gemm_xwT(const __bf16* __restrict__ A, const __bf16* __restrict__ Bt,
                 const float* __restrict__ bias, void* __restrict__ Cout,
                 int M, int N, int K) {
  const int wid = (blockIdx.x * blockDim.x + threadIdx.x) >> 5;
  const int bn = N >> 6;
  const int wm = wid / bn;
  const int wn = wid % bn;
  if (wm >= (M >> 6)) return;   // wave-uniform; grids are exact
  const int row0 = wm << 6, col0 = wn << 6;
  v8f acc[4][4];
#pragma unroll
  for (int i = 0; i < 4; ++i)
#pragma unroll
    for (int j = 0; j < 4; ++j) acc[i][j] = {0.f, 0.f, 0.f, 0.f, 0.f, 0.f, 0.f, 0.f};
  for (int k0 = 0; k0 < K; k0 += 32) {
    v16bf a[4], b[4];
#pragma unroll
    for (int i = 0; i < 4; ++i) a[i] = load_a_bf16(A, K, row0 + (i << 4), k0);
#pragma unroll
    for (int j = 0; j < 4; ++j) b[j] = load_b_bf16(Bt, K, col0 + (j << 4), k0);
#pragma unroll
    for (int i = 0; i < 4; ++i)
#pragma unroll
      for (int j = 0; j < 4; ++j) acc[i][j] = wmma_bf16(a[i], b[j], acc[i][j]);
  }
#pragma unroll
  for (int j = 0; j < 4; ++j) {
    const float bc = bias[col0 + (j << 4) + (threadIdx.x & 15)];
#pragma unroll
    for (int i = 0; i < 4; ++i) {
#pragma unroll
      for (int r = 0; r < 8; ++r) acc[i][j][r] += bc;
      if (OUT_BF16) store_c_bf16((__bf16*)Cout, N, row0 + (i << 4), col0 + (j << 4), acc[i][j]);
      else          store_c_f32((float*)Cout, N, row0 + (i << 4), col0 + (j << 4), acc[i][j]);
    }
  }
}

// Vt[b,h,d,m] = Vh[b, m, h*64+d]  (bf16)
__global__ void aggattn_vt(const __bf16* __restrict__ Vh, __bf16* __restrict__ Vt) {
  int idx = blockIdx.x * blockDim.x + threadIdx.x;     // B*H*HEAD_DIM*SEQ = 4194304
  int m  = idx & (SEQ - 1);
  int d  = (idx >> 10) & (HEAD_DIM - 1);
  int bh = idx >> 16;
  Vt[idx] = Vh[((size_t)(bh >> 4) * SEQ + m) * D_MODEL + (bh & 15) * HEAD_DIM + d];
}

// Pack concatenated score operands (per-head contiguous, ld = 128):
//   A2[bh,l,0:64]  = lam_cross*scale*Qh,  A2[bh,l,64:128] = lam_self*scale*Kh
//   B2[bh,l,0:64]  = Kh,                  B2[bh,l,64:128] = Vh
// so that scores = A2 @ B2^T in a single WMMA accumulation chain (K = 128).
__global__ void aggattn_pack(const __bf16* __restrict__ Qh, const __bf16* __restrict__ Kh,
                             const __bf16* __restrict__ Vh, const float* __restrict__ lam,
                             __bf16* __restrict__ A2, __bf16* __restrict__ B2) {
  int idx = blockIdx.x * blockDim.x + threadIdx.x;     // B*H*SEQ*128 = 8388608
  int d  = idx & 127;
  int l  = (idx >> 7) & (SEQ - 1);
  int bh = idx >> 17;
  size_t src = ((size_t)(bh >> 4) * SEQ + l) * D_MODEL + (bh & 15) * HEAD_DIM + (d & 63);
  const float lc_s = lam[1] * 0.125f;   // lam_cross / sqrt(64)
  const float ls_s = lam[0] * 0.125f;   // lam_self  / sqrt(64)
  if (d < 64) {
    A2[idx] = (__bf16)((float)Qh[src] * lc_s);
    B2[idx] = Kh[src];
  } else {
    A2[idx] = (__bf16)((float)Kh[src] * ls_s);
    B2[idx] = Vh[src];
  }
}

// scores = A2 @ B2^T per head; one wave -> 64x64 output, K = 128.
__global__ void __launch_bounds__(256, 1)
aggattn_scores(const __bf16* __restrict__ A2, const __bf16* __restrict__ B2,
               float* __restrict__ attn) {
  const int wid = (blockIdx.x * blockDim.x + threadIdx.x) >> 5;
  const int wn = wid & 15;
  const int wm = (wid >> 4) & 15;
  const int bh = wid >> 8;                 // 0..63
  const __bf16* ab = A2 + (size_t)bh * SEQ * 128;
  const __bf16* bb = B2 + (size_t)bh * SEQ * 128;
  const int row0 = wm << 6, col0 = wn << 6;
  v8f acc[4][4];
#pragma unroll
  for (int i = 0; i < 4; ++i)
#pragma unroll
    for (int j = 0; j < 4; ++j) acc[i][j] = {0.f, 0.f, 0.f, 0.f, 0.f, 0.f, 0.f, 0.f};
#pragma unroll
  for (int k0 = 0; k0 < 128; k0 += 32) {
    v16bf a[4], b[4];
#pragma unroll
    for (int i = 0; i < 4; ++i) a[i] = load_a_bf16(ab, 128, row0 + (i << 4), k0);
#pragma unroll
    for (int j = 0; j < 4; ++j) b[j] = load_b_bf16(bb, 128, col0 + (j << 4), k0);
#pragma unroll
    for (int i = 0; i < 4; ++i)
#pragma unroll
      for (int j = 0; j < 4; ++j) acc[i][j] = wmma_bf16(a[i], b[j], acc[i][j]);
  }
  float* cb = attn + (size_t)bh * SEQ * SEQ;
#pragma unroll
  for (int i = 0; i < 4; ++i)
#pragma unroll
    for (int j = 0; j < 4; ++j)
      store_c_f32(cb, SEQ, row0 + (i << 4), col0 + (j << 4), acc[i][j]);
}

// row softmax with nan_to_num, in place; one 256-thread block per row of 1024
__global__ void aggattn_softmax(float* __restrict__ attn) {
  __shared__ float red[256];
  const int t = threadIdx.x;
  float* row = attn + (size_t)blockIdx.x * SEQ;
  float v[4];
  float mx = -3.0e38f;
#pragma unroll
  for (int i = 0; i < 4; ++i) {
    float x = row[t + i * 256];
    x = (x != x) ? 0.0f : x;
    v[i] = x;
    mx = fmaxf(mx, x);
  }
  red[t] = mx; __syncthreads();
  for (int s = 128; s > 0; s >>= 1) { if (t < s) red[t] = fmaxf(red[t], red[t + s]); __syncthreads(); }
  mx = red[0]; __syncthreads();
  float sum = 0.f;
#pragma unroll
  for (int i = 0; i < 4; ++i) { v[i] = __expf(v[i] - mx); sum += v[i]; }
  red[t] = sum; __syncthreads();
  for (int s = 128; s > 0; s >>= 1) { if (t < s) red[t] += red[t + s]; __syncthreads(); }
  const float inv = 1.0f / red[0];
#pragma unroll
  for (int i = 0; i < 4; ++i) row[t + i * 256] = v[i] * inv;
}

// AO[b,l,h*64+d] = sum_m attn[bh,l,m] * V[bh,m,d]; one wave -> 64 rows x full head (64).
__global__ void __launch_bounds__(256, 1)
aggattn_av(const float* __restrict__ attn, const __bf16* __restrict__ Vt,
           __bf16* __restrict__ AO) {
  const int wid = (blockIdx.x * blockDim.x + threadIdx.x) >> 5;
  const int wm = wid & 15;
  const int bh = wid >> 4;                  // 0..63
  const float*  ab  = attn + (size_t)bh * SEQ * SEQ;          // ld = SEQ
  const __bf16* vtb = Vt + (size_t)bh * HEAD_DIM * SEQ;       // [64 x SEQ], ld = SEQ
  const int row0 = wm << 6;
  v8f acc[4][4];
#pragma unroll
  for (int i = 0; i < 4; ++i)
#pragma unroll
    for (int j = 0; j < 4; ++j) acc[i][j] = {0.f, 0.f, 0.f, 0.f, 0.f, 0.f, 0.f, 0.f};
  for (int k0 = 0; k0 < SEQ; k0 += 32) {
    v16bf a[4], b[4];
#pragma unroll
    for (int i = 0; i < 4; ++i) a[i] = load_a_f32(ab, SEQ, row0 + (i << 4), k0);
#pragma unroll
    for (int j = 0; j < 4; ++j) b[j] = load_b_bf16(vtb, SEQ, j << 4, k0);
#pragma unroll
    for (int i = 0; i < 4; ++i)
#pragma unroll
      for (int j = 0; j < 4; ++j) acc[i][j] = wmma_bf16(a[i], b[j], acc[i][j]);
  }
  __bf16* ob = AO + (size_t)(bh >> 4) * SEQ * D_MODEL + (size_t)(bh & 15) * HEAD_DIM;
#pragma unroll
  for (int i = 0; i < 4; ++i)
#pragma unroll
    for (int j = 0; j < 4; ++j)
      store_c_bf16(ob, D_MODEL, row0 + (i << 4), j << 4, acc[i][j]);
}

// ---------------- Host launch ----------------
extern "C" void kernel_launch(void* const* d_in, const int* in_sizes, int n_in,
                              void* d_out, int out_size, void* d_ws, size_t ws_size,
                              hipStream_t stream) {
  const float* q  = (const float*)d_in[0];
  const float* k  = (const float*)d_in[1];
  const float* v  = (const float*)d_in[2];
  const float* Wq = (const float*)d_in[3];
  const float* bq = (const float*)d_in[4];
  const float* Wk = (const float*)d_in[5];
  const float* bk = (const float*)d_in[6];
  const float* Wv = (const float*)d_in[7];
  const float* bv = (const float*)d_in[8];
  const float* Wo = (const float*)d_in[9];
  const float* bo = (const float*)d_in[10];
  const float* lq = (const float*)d_in[11];
  const float* lk = (const float*)d_in[12];
  const float* lv = (const float*)d_in[13];

  float* out  = (float*)d_out;                                   // [B, L, D]
  float* attn = out + (size_t)BATCH * SEQ * D_MODEL;             // [B, H, L, L]

  const size_t NX = (size_t)BATCH * SEQ * D_MODEL;               // 4194304
  const size_t NW = (size_t)D_MODEL * D_MODEL;                   // 1048576
  const size_t N2 = (size_t)BATCH * NUM_HEADS * SEQ * 128;       // 8388608

  float*  lam  = (float*)d_ws;
  __bf16* base = (__bf16*)((char*)d_ws + 256);
  __bf16* qb  = base;
  __bf16* kb  = base + NX;
  __bf16* vb  = base + 2 * NX;
  __bf16* wqb = base + 3 * NX;
  __bf16* wkb = wqb + NW;
  __bf16* wvb = wqb + 2 * NW;
  __bf16* wob = wqb + 3 * NW;
  __bf16* Qh  = wqb + 4 * NW;
  __bf16* Kh  = Qh + NX;
  __bf16* Vh  = Qh + 2 * NX;
  __bf16* Vt  = Qh + 3 * NX;
  __bf16* AO  = Qh + 4 * NX;
  __bf16* A2  = Qh + 5 * NX;
  __bf16* B2  = A2 + N2;       // total ws: (8*NX + 4*NW + 2*N2)*2 + 256 ~= 109 MB

  // 1. lambda scalars
  aggattn_lambda<<<1, 32, 0, stream>>>(lq, lk, lv, lam);

  // 2. f32 -> bf16 conversions
  aggattn_cvt<<<(int)(NX / 256), 256, 0, stream>>>(q, qb, (int)NX);
  aggattn_cvt<<<(int)(NX / 256), 256, 0, stream>>>(k, kb, (int)NX);
  aggattn_cvt<<<(int)(NX / 256), 256, 0, stream>>>(v, vb, (int)NX);
  aggattn_cvt<<<(int)(NW / 256), 256, 0, stream>>>(Wq, wqb, (int)NW);
  aggattn_cvt<<<(int)(NW / 256), 256, 0, stream>>>(Wk, wkb, (int)NW);
  aggattn_cvt<<<(int)(NW / 256), 256, 0, stream>>>(Wv, wvb, (int)NW);
  aggattn_cvt<<<(int)(NW / 256), 256, 0, stream>>>(Wo, wob, (int)NW);

  // 3. projections: [4096x1024] @ W^T + b -> bf16. 64x64 per wave:
  //    waves = (4096/64)*(1024/64) = 1024 -> 128 blocks of 8 waves
  const int M = BATCH * SEQ;
  const int gemmBlocks = (M / 64) * (D_MODEL / 64) / 8;   // 128
  aggattn_gemm_xwT<true><<<gemmBlocks, 256, 0, stream>>>(qb, wqb, bq, Qh, M, D_MODEL, D_MODEL);
  aggattn_gemm_xwT<true><<<gemmBlocks, 256, 0, stream>>>(kb, wkb, bk, Kh, M, D_MODEL, D_MODEL);
  aggattn_gemm_xwT<true><<<gemmBlocks, 256, 0, stream>>>(vb, wvb, bv, Vh, M, D_MODEL, D_MODEL);

  // 4. per-head transposed V copy + concatenated score operands
  aggattn_vt<<<(int)(NX / 256), 256, 0, stream>>>(Vh, Vt);
  aggattn_pack<<<(int)(N2 / 256), 256, 0, stream>>>(Qh, Kh, Vh, lam, A2, B2);

  // 5. fused scores -> attn region of d_out (f32); waves = 64*256 = 16384 -> 2048 blocks
  aggattn_scores<<<2048, 256, 0, stream>>>(A2, B2, attn);

  // 6. softmax in place, one block per row
  aggattn_softmax<<<BATCH * NUM_HEADS * SEQ, 256, 0, stream>>>(attn);

  // 7. attn @ V -> AO (bf16, head-merged layout); waves = 64*16 = 1024 -> 128 blocks
  aggattn_av<<<128, 256, 0, stream>>>(attn, Vt, AO);

  // 8. output projection -> f32 out
  aggattn_gemm_xwT<false><<<gemmBlocks, 256, 0, stream>>>(AO, wob, bo, out, M, D_MODEL, D_MODEL);
}
